// DecoderRNN_32719060860943
// MI455X (gfx1250) — compile-verified, matrix operand order checked by reading, not run on hardware
//
#include <hip/hip_runtime.h>
#include <math.h>

// ---------------------------------------------------------------------------
// DecoderRNN inference for MI455X (gfx1250, wave32, WMMA).
// All GEMMs run through v_wmma_f32_16x16x32_bf16 with bf16 weights converted
// once into workspace. Every GEMM in this model has M == 256, so each
// 8-wave block covers the full M (waves stacked in M) sharing one 64-wide
// N-strip: the 8 waves hit the same weight cachelines near-simultaneously,
// so weights stream from L2 once per block and re-serve from WGP$ (8x less
// L2 weight traffic on the dominant 32000-wide vocab GEMM). Per-wave tile is
// 32x64 (8 WMMA accumulators, 1.5 b128 loads per WMMA).
// ---------------------------------------------------------------------------

typedef __bf16 bf16_t;
typedef __attribute__((ext_vector_type(8)))  bf16_t v8bf;
typedef __attribute__((ext_vector_type(16))) bf16_t v16bf;
typedef __attribute__((ext_vector_type(8)))  float  v8f;

#define B_  256
#define H_  1024
#define E_  18432
#define V_  32000
#define T_  15
#define G_  (4 * H_)

// ------------------------- fragment loader ---------------------------------
// 16-bit A/B fragment, wave32 (ISA 7.12.2): lanes 0-15 hold rows 0-15 with
// K = {k..k+7, k+16..k+23}; lanes 16-31 hold the same rows with
// K = {k+8..k+15, k+24..k+31}.  Both x[M,K] and W[N,K] are K-contiguous, so
// each lane does two 16-byte loads.
__device__ __forceinline__ v16bf load_frag(const bf16_t* base, int ldk, int kk, int lane)
{
    const bf16_t* p = base + (size_t)(lane & 15) * (size_t)ldk
                           + (size_t)(kk + ((lane >> 4) << 3));
    v8bf lo = *(const v8bf*)(p);
    v8bf hi = *(const v8bf*)(p + 16);
    return __builtin_shufflevector(lo, hi, 0, 1, 2, 3, 4, 5, 6, 7,
                                            8, 9, 10, 11, 12, 13, 14, 15);
}

// C/D layout (ISA 7.12.2): lane's column n = lane&15; VGPR r holds row
// r + (lane>=16 ? 8 : 0).
__device__ __forceinline__ void store_tile(v8f acc, int mbase, int nbase,
                                           const float* bias1, const float* bias2,
                                           float* outF, bf16_t* outB, int N, int relu,
                                           int lane)
{
    int n  = nbase + (lane & 15);
    int mo = (lane >> 4) << 3;
    float bv = 0.0f;
    if (bias1) bv += bias1[n];
    if (bias2) bv += bias2[n];
#pragma unroll
    for (int r = 0; r < 8; ++r) {
        float v = acc[r] + bv;
        if (relu) v = fmaxf(v, 0.0f);
        size_t off = (size_t)(mbase + mo + r) * (size_t)N + (size_t)n;
        if (outF) outF[off] = v;
        if (outB) outB[off] = (bf16_t)v;
    }
}

// ------------------------- WMMA GEMM ---------------------------------------
// out[256,N] = A[256,K1] @ W[N,K1]^T (+ A2[256,K2] @ W2[N,K2]^T) (+ biases).
// Block = 8 waves stacked in M (tm = wave*32, covers M=256 exactly);
// blockIdx.x selects a 64-wide N strip. Per-wave tile 32x64 -> acc[2][4].
__global__ void __launch_bounds__(256)
gemm_bf16_wmma(const bf16_t* __restrict__ A,  const bf16_t* __restrict__ W,  int K1,
               const bf16_t* __restrict__ A2, const bf16_t* __restrict__ W2, int K2,
               const float* __restrict__ bias1, const float* __restrict__ bias2,
               float* __restrict__ outF, bf16_t* __restrict__ outB,
               int N, int relu)
{
    const int lane = threadIdx.x & 31;
    const int wave = threadIdx.x >> 5;
    const int tm = wave << 5;        // waves 0..7 cover rows 0..255
    const int tn = blockIdx.x << 6;  // 64-wide N strip per block

    v8f acc[2][4] = {};

    {
        const bf16_t* Ar0 = A + (size_t)tm * K1;
        const bf16_t* Ar1 = A + (size_t)(tm + 16) * K1;
        const bf16_t* Wr0 = W + (size_t)(tn)      * K1;
        const bf16_t* Wr1 = W + (size_t)(tn + 16) * K1;
        const bf16_t* Wr2 = W + (size_t)(tn + 32) * K1;
        const bf16_t* Wr3 = W + (size_t)(tn + 48) * K1;
        for (int k = 0; k < K1; k += 32) {
            v16bf a0 = load_frag(Ar0, K1, k, lane);
            v16bf a1 = load_frag(Ar1, K1, k, lane);
            v16bf b0 = load_frag(Wr0, K1, k, lane);
            v16bf b1 = load_frag(Wr1, K1, k, lane);
            v16bf b2 = load_frag(Wr2, K1, k, lane);
            v16bf b3 = load_frag(Wr3, K1, k, lane);
            acc[0][0] = __builtin_amdgcn_wmma_f32_16x16x32_bf16(false, a0, false, b0, (short)0, acc[0][0], false, false);
            acc[0][1] = __builtin_amdgcn_wmma_f32_16x16x32_bf16(false, a0, false, b1, (short)0, acc[0][1], false, false);
            acc[0][2] = __builtin_amdgcn_wmma_f32_16x16x32_bf16(false, a0, false, b2, (short)0, acc[0][2], false, false);
            acc[0][3] = __builtin_amdgcn_wmma_f32_16x16x32_bf16(false, a0, false, b3, (short)0, acc[0][3], false, false);
            acc[1][0] = __builtin_amdgcn_wmma_f32_16x16x32_bf16(false, a1, false, b0, (short)0, acc[1][0], false, false);
            acc[1][1] = __builtin_amdgcn_wmma_f32_16x16x32_bf16(false, a1, false, b1, (short)0, acc[1][1], false, false);
            acc[1][2] = __builtin_amdgcn_wmma_f32_16x16x32_bf16(false, a1, false, b2, (short)0, acc[1][2], false, false);
            acc[1][3] = __builtin_amdgcn_wmma_f32_16x16x32_bf16(false, a1, false, b3, (short)0, acc[1][3], false, false);
        }
    }
    if (A2) {  // fused second product: x@Wih^T + h@Whh^T for LSTM gates
        const bf16_t* Ar0 = A2 + (size_t)tm * K2;
        const bf16_t* Ar1 = A2 + (size_t)(tm + 16) * K2;
        const bf16_t* Wr0 = W2 + (size_t)(tn)      * K2;
        const bf16_t* Wr1 = W2 + (size_t)(tn + 16) * K2;
        const bf16_t* Wr2 = W2 + (size_t)(tn + 32) * K2;
        const bf16_t* Wr3 = W2 + (size_t)(tn + 48) * K2;
        for (int k = 0; k < K2; k += 32) {
            v16bf a0 = load_frag(Ar0, K2, k, lane);
            v16bf a1 = load_frag(Ar1, K2, k, lane);
            v16bf b0 = load_frag(Wr0, K2, k, lane);
            v16bf b1 = load_frag(Wr1, K2, k, lane);
            v16bf b2 = load_frag(Wr2, K2, k, lane);
            v16bf b3 = load_frag(Wr3, K2, k, lane);
            acc[0][0] = __builtin_amdgcn_wmma_f32_16x16x32_bf16(false, a0, false, b0, (short)0, acc[0][0], false, false);
            acc[0][1] = __builtin_amdgcn_wmma_f32_16x16x32_bf16(false, a0, false, b1, (short)0, acc[0][1], false, false);
            acc[0][2] = __builtin_amdgcn_wmma_f32_16x16x32_bf16(false, a0, false, b2, (short)0, acc[0][2], false, false);
            acc[0][3] = __builtin_amdgcn_wmma_f32_16x16x32_bf16(false, a0, false, b3, (short)0, acc[0][3], false, false);
            acc[1][0] = __builtin_amdgcn_wmma_f32_16x16x32_bf16(false, a1, false, b0, (short)0, acc[1][0], false, false);
            acc[1][1] = __builtin_amdgcn_wmma_f32_16x16x32_bf16(false, a1, false, b1, (short)0, acc[1][1], false, false);
            acc[1][2] = __builtin_amdgcn_wmma_f32_16x16x32_bf16(false, a1, false, b2, (short)0, acc[1][2], false, false);
            acc[1][3] = __builtin_amdgcn_wmma_f32_16x16x32_bf16(false, a1, false, b3, (short)0, acc[1][3], false, false);
        }
    }

#pragma unroll
    for (int i = 0; i < 2; ++i)
#pragma unroll
        for (int j = 0; j < 4; ++j)
            store_tile(acc[i][j], tm + 16 * i, tn + 16 * j,
                       bias1, bias2, outF, outB, N, relu, lane);
}

// ------------------------- elementwise kernels -----------------------------
__global__ void f32_to_bf16_kernel(const float* __restrict__ src,
                                   bf16_t* __restrict__ dst, size_t n)
{
    size_t i = (size_t)blockIdx.x * blockDim.x + threadIdx.x;
    if (i < n) dst[i] = (bf16_t)src[i];
}

// LayerNorm over H, optional residual add, write f32 and/or bf16.
__global__ void __launch_bounds__(256)
layernorm_kernel(const float* __restrict__ x, const float* __restrict__ g,
                 const float* __restrict__ b, const float* __restrict__ add,
                 float* __restrict__ outF, bf16_t* __restrict__ outB, int H)
{
    __shared__ float s1[256];
    __shared__ float s2[256];
    const int row = blockIdx.x;
    const float* xr = x + (size_t)row * H;
    float a1 = 0.0f, a2 = 0.0f;
    for (int i = threadIdx.x; i < H; i += 256) {
        float v = xr[i];
        a1 += v; a2 += v * v;
    }
    s1[threadIdx.x] = a1; s2[threadIdx.x] = a2;
    __syncthreads();
    for (int s = 128; s > 0; s >>= 1) {
        if ((int)threadIdx.x < s) {
            s1[threadIdx.x] += s1[threadIdx.x + s];
            s2[threadIdx.x] += s2[threadIdx.x + s];
        }
        __syncthreads();
    }
    const float mean = s1[0] / (float)H;
    const float var  = s2[0] / (float)H - mean * mean;
    const float inv  = rsqrtf(var + 1e-5f);
    for (int i = threadIdx.x; i < H; i += 256) {
        float v = (xr[i] - mean) * inv * g[i] + b[i];
        if (add) v += add[(size_t)row * H + i];
        size_t off = (size_t)row * H + i;
        if (outF) outF[off] = v;
        if (outB) outB[off] = (bf16_t)v;
    }
}

// LSTM gate activation: gates[B,4H] (i,f,g,o) + cprev -> h (f32+bf16), c.
__global__ void lstm_act_kernel(const float* __restrict__ gates,
                                const float* __restrict__ cprev,
                                float* __restrict__ hF, bf16_t* __restrict__ hB,
                                float* __restrict__ cF, int H)
{
    int idx = blockIdx.x * blockDim.x + threadIdx.x;
    if (idx >= B_ * H) return;
    int b = idx / H, j = idx - b * H;
    const float* gr = gates + (size_t)b * 4 * H;
    float i_ = 1.0f / (1.0f + expf(-gr[j]));
    float f_ = 1.0f / (1.0f + expf(-gr[H + j]));
    float g_ = tanhf(gr[2 * H + j]);
    float o_ = 1.0f / (1.0f + expf(-gr[3 * H + j]));
    float c  = f_ * cprev[idx] + i_ * g_;
    float h  = o_ * tanhf(c);
    cF[idx] = c;
    hF[idx] = h;
    hB[idx] = (bf16_t)h;
}

// Row-wise argmax over V (first-max semantics).
__global__ void __launch_bounds__(256)
argmax_kernel(const float* __restrict__ pred, int* __restrict__ idx, int V)
{
    __shared__ float sv[256];
    __shared__ int   si[256];
    const int row = blockIdx.x;
    const float* pr = pred + (size_t)row * V;
    float best = -INFINITY; int bi = V;
    for (int i = threadIdx.x; i < V; i += 256) {
        float v = pr[i];
        if (v > best) { best = v; bi = i; }
    }
    sv[threadIdx.x] = best; si[threadIdx.x] = bi;
    __syncthreads();
    for (int s = 128; s > 0; s >>= 1) {
        if ((int)threadIdx.x < s) {
            float vo = sv[threadIdx.x + s]; int io = si[threadIdx.x + s];
            if (vo > sv[threadIdx.x] ||
                (vo == sv[threadIdx.x] && io < si[threadIdx.x])) {
                sv[threadIdx.x] = vo; si[threadIdx.x] = io;
            }
        }
        __syncthreads();
    }
    if (threadIdx.x == 0) idx[row] = si[0];
}

// out column t = totoken_w[idx[0]] broadcast over batch; word = embed[idx[b]].
__global__ void token_embed_kernel(const int* __restrict__ idx,
                                   const float* __restrict__ totoken,
                                   const bf16_t* __restrict__ embB,
                                   bf16_t* __restrict__ wordB,
                                   float* __restrict__ out, int t)
{
    int i = blockIdx.x * blockDim.x + threadIdx.x;
    if (i < B_) out[(size_t)i * T_ + t] = totoken[idx[0]];
    if (i < B_ * H_) {
        int b = i / H_, j = i - b * H_;
        wordB[i] = embB[(size_t)idx[b] * H_ + j];
    }
}

// ------------------------- host-side helpers -------------------------------
static inline char* bump(char*& p, size_t bytes)
{
    char* r = p;
    p += (bytes + 255) & ~(size_t)255;
    return r;
}

static inline void conv_bf16(const float* s, bf16_t* d, size_t n, hipStream_t st)
{
    f32_to_bf16_kernel<<<(unsigned)((n + 255) / 256), 256, 0, st>>>(s, d, n);
}

static inline void run_gemm(const bf16_t* A, const bf16_t* W, int K1,
                            const bf16_t* A2, const bf16_t* W2, int K2,
                            const float* b1, const float* b2,
                            float* oF, bf16_t* oB, int N, int relu,
                            hipStream_t st)
{
    // M is always 256 (one 8-wave block spans all of M); N % 64 == 0.
    gemm_bf16_wmma<<<N / 64, 256, 0, st>>>(A, W, K1, A2, W2, K2,
                                           b1, b2, oF, oB, N, relu);
}

extern "C" void kernel_launch(void* const* d_in, const int* in_sizes, int n_in,
                              void* d_out, int out_size, void* d_ws, size_t ws_size,
                              hipStream_t stream)
{
    (void)in_sizes; (void)n_in; (void)out_size; (void)ws_size;

    // ---- inputs (setup_inputs order) ----
    const float* features = (const float*)d_in[0];
    const float* fc1_w    = (const float*)d_in[1];
    const float* fc1_b    = (const float*)d_in[2];
    const float* ih_w     = (const float*)d_in[3];
    const float* ih_b     = (const float*)d_in[4];
    const float* ic_w     = (const float*)d_in[5];
    const float* ic_b     = (const float*)d_in[6];
    const float* ln_h_g   = (const float*)d_in[7];
    const float* ln_h_b   = (const float*)d_in[8];
    const float* ln_c_g   = (const float*)d_in[9];
    const float* ln_c_b   = (const float*)d_in[10];
    const float* w_ih1    = (const float*)d_in[11];
    const float* w_hh1    = (const float*)d_in[12];
    const float* b_ih1    = (const float*)d_in[13];
    const float* b_hh1    = (const float*)d_in[14];
    const float* w_ih2    = (const float*)d_in[15];
    const float* w_hh2    = (const float*)d_in[16];
    const float* b_ih2    = (const float*)d_in[17];
    const float* b_hh2    = (const float*)d_in[18];
    const float* fc_w     = (const float*)d_in[19];
    const float* fc_b     = (const float*)d_in[20];
    const float* embed_w  = (const float*)d_in[21];
    const float* totoken  = (const float*)d_in[22];
    // d_in[23] = max_length (device scalar). Graph capture forbids a sync to
    // read it; it is 15 by construction of setup_inputs -> T_ == 15.
    float* out = (float*)d_out;

    // ---- workspace layout (bump allocation, ~250 MB) ----
    char* p = (char*)d_ws;
    bf16_t* featbf = (bf16_t*)bump(p, (size_t)B_ * E_ * 2);
    bf16_t* fc1bw  = (bf16_t*)bump(p, (size_t)H_ * E_ * 2);
    bf16_t* ihbw   = (bf16_t*)bump(p, (size_t)H_ * H_ * 2);
    bf16_t* icbw   = (bf16_t*)bump(p, (size_t)H_ * H_ * 2);
    bf16_t* wih1b  = (bf16_t*)bump(p, (size_t)G_ * H_ * 2);
    bf16_t* whh1b  = (bf16_t*)bump(p, (size_t)G_ * H_ * 2);
    bf16_t* wih2b  = (bf16_t*)bump(p, (size_t)G_ * H_ * 2);
    bf16_t* whh2b  = (bf16_t*)bump(p, (size_t)G_ * H_ * 2);
    bf16_t* fcwb   = (bf16_t*)bump(p, (size_t)V_ * H_ * 2);
    bf16_t* embb   = (bf16_t*)bump(p, (size_t)V_ * H_ * 2);
    float*  featF  = (float*)bump(p, (size_t)B_ * H_ * 4);
    bf16_t* featB  = (bf16_t*)bump(p, (size_t)B_ * H_ * 2);
    float*  tmpA   = (float*)bump(p, (size_t)B_ * H_ * 4);
    float*  hF     = (float*)bump(p, (size_t)B_ * H_ * 4);
    bf16_t* hB     = (bf16_t*)bump(p, (size_t)B_ * H_ * 2);
    float*  cF     = (float*)bump(p, (size_t)B_ * H_ * 4);
    float*  h1F    = (float*)bump(p, (size_t)B_ * H_ * 4);
    bf16_t* h1B    = (bf16_t*)bump(p, (size_t)B_ * H_ * 2);
    float*  c1F    = (float*)bump(p, (size_t)B_ * H_ * 4);
    float*  h2F    = (float*)bump(p, (size_t)B_ * H_ * 4);
    bf16_t* h2B    = (bf16_t*)bump(p, (size_t)B_ * H_ * 2);
    float*  c2F    = (float*)bump(p, (size_t)B_ * H_ * 4);
    float*  h3F    = (float*)bump(p, (size_t)B_ * H_ * 4);
    bf16_t* h3B    = (bf16_t*)bump(p, (size_t)B_ * H_ * 2);
    float*  c3F    = (float*)bump(p, (size_t)B_ * H_ * 4);
    bf16_t* lnAB   = (bf16_t*)bump(p, (size_t)B_ * H_ * 2);
    float*  lnCF   = (float*)bump(p, (size_t)B_ * H_ * 4);
    float*  gates  = (float*)bump(p, (size_t)B_ * G_ * 4);
    float*  pred   = (float*)bump(p, (size_t)B_ * V_ * 4);
    bf16_t* wordB  = (bf16_t*)bump(p, (size_t)B_ * H_ * 2);
    int*    idx    = (int*)bump(p, (size_t)B_ * 4);

    const int NBH = (B_ * H_ + 255) / 256;

    // ---- one-time bf16 conversions (reused across all 15 steps) ----
    conv_bf16(features, featbf, (size_t)B_ * E_, stream);
    conv_bf16(fc1_w,    fc1bw,  (size_t)H_ * E_, stream);
    conv_bf16(ih_w,     ihbw,   (size_t)H_ * H_, stream);
    conv_bf16(ic_w,     icbw,   (size_t)H_ * H_, stream);
    conv_bf16(w_ih1,    wih1b,  (size_t)G_ * H_, stream);
    conv_bf16(w_hh1,    whh1b,  (size_t)G_ * H_, stream);
    conv_bf16(w_ih2,    wih2b,  (size_t)G_ * H_, stream);
    conv_bf16(w_hh2,    whh2b,  (size_t)G_ * H_, stream);
    conv_bf16(fc_w,     fcwb,   (size_t)V_ * H_, stream);
    conv_bf16(embed_w,  embb,   (size_t)V_ * H_, stream);

    // ---- feat = relu(features @ fc1_w^T + fc1_b) ----
    run_gemm(featbf, fc1bw, E_, nullptr, nullptr, 0, fc1_b, nullptr,
             featF, featB, H_, 1, stream);

    // ---- h = LN(relu(feat @ ih_w^T + ih_b)); c likewise ----
    run_gemm(featB, ihbw, H_, nullptr, nullptr, 0, ih_b, nullptr,
             tmpA, nullptr, H_, 1, stream);
    layernorm_kernel<<<B_, 256, 0, stream>>>(tmpA, ln_h_g, ln_h_b, nullptr, hF, hB, H_);
    run_gemm(featB, icbw, H_, nullptr, nullptr, 0, ic_b, nullptr,
             tmpA, nullptr, H_, 1, stream);
    layernorm_kernel<<<B_, 256, 0, stream>>>(tmpA, ln_c_g, ln_c_b, nullptr, cF, nullptr, H_);

    // ---- step 0: three stacked LSTM cells, all seeded with (h, c) ----
    run_gemm(featB, wih1b, H_, hB, whh1b, H_, b_ih1, b_hh1,
             gates, nullptr, G_, 0, stream);
    lstm_act_kernel<<<NBH, 256, 0, stream>>>(gates, cF, h1F, h1B, c1F, H_);

    run_gemm(h1B, wih2b, H_, hB, whh2b, H_, b_ih2, b_hh2,
             gates, nullptr, G_, 0, stream);
    lstm_act_kernel<<<NBH, 256, 0, stream>>>(gates, cF, h2F, h2B, c2F, H_);

    run_gemm(h2B, wih2b, H_, hB, whh2b, H_, b_ih2, b_hh2,
             gates, nullptr, G_, 0, stream);
    lstm_act_kernel<<<NBH, 256, 0, stream>>>(gates, cF, h3F, h3B, c3F, H_);

    run_gemm(h3B, fcwb, H_, nullptr, nullptr, 0, fc_b, nullptr,
             pred, nullptr, V_, 0, stream);
    argmax_kernel<<<B_, 256, 0, stream>>>(pred, idx, V_);
    token_embed_kernel<<<NBH, 256, 0, stream>>>(idx, totoken, embb, wordB, out, 0);

    // ---- decode steps 1..T-1 ----
    for (int t = 1; t < T_; ++t) {
        layernorm_kernel<<<B_, 256, 0, stream>>>(h1F, ln_h_g, ln_h_b, hF, nullptr, lnAB, H_);
        layernorm_kernel<<<B_, 256, 0, stream>>>(c1F, ln_c_g, ln_c_b, cF, lnCF, nullptr, H_);

        run_gemm(wordB, wih1b, H_, lnAB, whh1b, H_, b_ih1, b_hh1,
                 gates, nullptr, G_, 0, stream);
        lstm_act_kernel<<<NBH, 256, 0, stream>>>(gates, lnCF, h1F, h1B, c1F, H_);

        run_gemm(h1B, wih2b, H_, h2B, whh2b, H_, b_ih2, b_hh2,
                 gates, nullptr, G_, 0, stream);
        lstm_act_kernel<<<NBH, 256, 0, stream>>>(gates, c2F, h2F, h2B, c2F, H_);

        run_gemm(h2B, wih2b, H_, h3B, whh2b, H_, b_ih2, b_hh2,
                 gates, nullptr, G_, 0, stream);
        lstm_act_kernel<<<NBH, 256, 0, stream>>>(gates, c3F, h3F, h3B, c3F, H_);

        run_gemm(h3B, fcwb, H_, nullptr, nullptr, 0, fc_b, nullptr,
                 pred, nullptr, V_, 0, stream);
        argmax_kernel<<<B_, 256, 0, stream>>>(pred, idx, V_);
        token_embed_kernel<<<NBH, 256, 0, stream>>>(idx, totoken, embb, wordB, out, t);
    }
}
